// _UnimolLayer_38070590112225
// MI455X (gfx1250) — matrix operand order repaired and checked by reading, not compiled, not run on hardware
//
#include <hip/hip_runtime.h>

// ---------------------------------------------------------------------------
// UnimolLayer for MI455X (gfx1250): wave32, WMMA f32_16x16x32_f16 everywhere.
// Sizes: L=1024, B=4, D=512, H=8, Dh=64, DFF=2048.  M = L*B = 4096 rows.
// Roofline: ~34 GFLOP compute, ~270 MB mandatory traffic (edge in + t+edge out)
// -> fuse edge into the attention score pass, do everything else in f16 WMMA
// with f32 accumulation. GEMMs: K-step 64, register double-buffered staging,
// global_prefetch for tile k+2.
// ---------------------------------------------------------------------------

typedef __attribute__((ext_vector_type(16))) _Float16 v16h;
typedef __attribute__((ext_vector_type(8)))  float    v8f;

#define WMMA_F16(a, b, c) \
    __builtin_amdgcn_wmma_f32_16x16x32_f16(false, (a), false, (b), (short)0, (c), false, false)

union Frag16 { v16h v; uint4 u[2]; };

// A-fragment (16x32 f16). ISA layout: lane = 16*khalf + row.
// VGPR0..3 : K = khalf*8 + 0..7 (2 f16/VGPR)   -> one b128
// VGPR4..7 : K = 16 + khalf*8 + 0..7           -> one b128
__device__ __forceinline__ v16h load_frag_a(const _Float16* s, int ld, int lane) {
    int r = lane & 15, kh = lane >> 4;
    const _Float16* p = s + r * ld;
    Frag16 f;
    f.u[0] = *(const uint4*)(p + kh * 8);
    f.u[1] = *(const uint4*)(p + 16 + kh * 8);
    return f.v;
}

// B-fragment (32x16 f16), tile stored as [N][K] (i.e. W row-major / K rows):
// same K order as A: elems 0..7 K = khalf*8 + 0..7, elems 8..15 K = 16 + khalf*8 + 0..7.
__device__ __forceinline__ v16h load_frag_b(const _Float16* s, int ld, int lane) {
    int c = lane & 15, kh = lane >> 4;
    const _Float16* p = s + c * ld;
    Frag16 f;
    f.u[0] = *(const uint4*)(p + kh * 8);
    f.u[1] = *(const uint4*)(p + 16 + kh * 8);
    return f.v;
}

// ---------------------------------------------------------------------------
// f32 -> f16 weight conversion
// ---------------------------------------------------------------------------
__global__ void k_f32_to_f16(const float* __restrict__ in, _Float16* __restrict__ out, int n) {
    int i = blockIdx.x * 256 + threadIdx.x;
    if (i < n) out[i] = (_Float16)in[i];
}

// ---------------------------------------------------------------------------
// LayerNorm over D=512, one row per block (256 threads), f16 output.
// ---------------------------------------------------------------------------
__global__ __launch_bounds__(256)
void k_layernorm_f16(const float* __restrict__ x, const float* __restrict__ w,
                     const float* __restrict__ b, _Float16* __restrict__ y) {
    __shared__ float red[2][8];
    int row = blockIdx.x, tid = threadIdx.x;
    const float* xr = x + (size_t)row * 512;
    float v0 = xr[tid], v1 = xr[tid + 256];
    float s = v0 + v1, ss = v0 * v0 + v1 * v1;
    for (int off = 16; off; off >>= 1) {
        s  += __shfl_xor(s,  off, 32);
        ss += __shfl_xor(ss, off, 32);
    }
    int wave = tid >> 5, lane = tid & 31;
    if (lane == 0) { red[0][wave] = s; red[1][wave] = ss; }
    __syncthreads();
    float S = 0.f, SS = 0.f;
    #pragma unroll
    for (int i = 0; i < 8; ++i) { S += red[0][i]; SS += red[1][i]; }
    float mu   = S * (1.f / 512.f);
    float var  = SS * (1.f / 512.f) - mu * mu;
    float rstd = rsqrtf(var + 1e-5f);
    y[(size_t)row * 512 + tid]       = (_Float16)((v0 - mu) * rstd * w[tid]       + b[tid]);
    y[(size_t)row * 512 + tid + 256] = (_Float16)((v1 - mu) * rstd * w[tid + 256] + b[tid + 256]);
}

// ---------------------------------------------------------------------------
// WMMA GEMM: C[M,N] = A[M,K] @ W[N,K]^T + bias (+res) (gelu?) ; A,W f16.
// 256 threads = 8 waves, 128x128 tile, wave grid 4x2, each wave 32x64.
// K-step 64, register double buffering: next tile's global loads are issued
// before the current tile's 16 WMMAs, hiding HBM/L2 latency. Prefetch k+2.
// ---------------------------------------------------------------------------
template <bool OUT_HALF, bool GELU, bool HAS_RES>
__global__ __launch_bounds__(256)
void k_gemm_wmma(const _Float16* __restrict__ A, const _Float16* __restrict__ W,
                 const float* __restrict__ bias, const float* __restrict__ res,
                 float* __restrict__ Cf, _Float16* __restrict__ Ch,
                 int M, int N, int K) {
    constexpr int KSTEP = 64;
    constexpr int LDT   = 72;                     // 64 + 8 halfs pad (144 B rows)
    __shared__ __align__(16) _Float16 As[128 * LDT];
    __shared__ __align__(16) _Float16 Bs[128 * LDT];
    int tid = threadIdx.x, lane = tid & 31, wave = tid >> 5;
    int wm = wave >> 1, wn = wave & 1;
    int bm = blockIdx.x * 128, bn = blockIdx.y * 128;

    int r = tid >> 1, cseg = (tid & 1) * 32;      // 2 threads per row, 32 halfs each
    const _Float16* gA = A + (size_t)(bm + r) * K + cseg;
    const _Float16* gW = W + (size_t)(bn + r) * K + cseg;

    uint4 ra[4], rb[4];
    auto gload = [&](int k0) {
        const uint4* pa = (const uint4*)(gA + k0);
        const uint4* pb = (const uint4*)(gW + k0);
        ra[0] = pa[0]; ra[1] = pa[1]; ra[2] = pa[2]; ra[3] = pa[3];
        rb[0] = pb[0]; rb[1] = pb[1]; rb[2] = pb[2]; rb[3] = pb[3];
    };
    auto sstore = [&]() {
        uint4* sa = (uint4*)(As + r * LDT + cseg);
        uint4* sb = (uint4*)(Bs + r * LDT + cseg);
        sa[0] = ra[0]; sa[1] = ra[1]; sa[2] = ra[2]; sa[3] = ra[3];
        sb[0] = rb[0]; sb[1] = rb[1]; sb[2] = rb[2]; sb[3] = rb[3];
    };

    v8f acc[2][4] = {};

    gload(0);
    sstore();
    for (int k0 = 0; k0 < K; k0 += KSTEP) {
        __syncthreads();                          // staged tile visible
        bool more = (k0 + KSTEP) < K;
        if (more) {
            gload(k0 + KSTEP);                    // in flight during WMMAs
            if (k0 + 2 * KSTEP < K) {             // pull k+2 into GL2 early
                __builtin_prefetch(gA + k0 + 2 * KSTEP, 0, 3);
                __builtin_prefetch(gW + k0 + 2 * KSTEP, 0, 3);
            }
        }
        #pragma unroll
        for (int ks = 0; ks < KSTEP; ks += 32) {
            v16h a0 = load_frag_a(As + (wm * 32 + 0)  * LDT + ks, LDT, lane);
            v16h a1 = load_frag_a(As + (wm * 32 + 16) * LDT + ks, LDT, lane);
            #pragma unroll
            for (int f = 0; f < 4; ++f) {
                v16h b = load_frag_b(Bs + (wn * 64 + f * 16) * LDT + ks, LDT, lane);
                acc[0][f] = WMMA_F16(a0, b, acc[0][f]);
                acc[1][f] = WMMA_F16(a1, b, acc[1][f]);
            }
        }
        __syncthreads();                          // all LDS reads done
        if (more) sstore();
    }

    int half = lane >> 4, c = lane & 15;
    #pragma unroll
    for (int mf = 0; mf < 2; ++mf)
        #pragma unroll
        for (int f = 0; f < 4; ++f) {
            int col = bn + wn * 64 + f * 16 + c;
            float bv = bias[col];
            #pragma unroll
            for (int v = 0; v < 8; ++v) {
                int row = bm + wm * 32 + mf * 16 + v + 8 * half;
                float val = acc[mf][f][v] + bv;
                if (HAS_RES) val += res[(size_t)row * N + col];
                if (GELU)    val = 0.5f * val * (1.f + erff(val * 0.70710678f));
                if (OUT_HALF) Ch[(size_t)row * N + col] = (_Float16)val;
                else          Cf[(size_t)row * N + col] = val;
            }
        }
}

// ---------------------------------------------------------------------------
// Fused attention + edge emit. Grid (32 heads, 16 q-blocks), 128 thr = 4 waves.
// Each wave owns 16 q-rows; flash-style loop over 16 k-tiles of 64.
//   S = (Q K^T)*0.125 + edge  ->  edge_out, online softmax, O += P V.
// ---------------------------------------------------------------------------
__global__ __launch_bounds__(128)
void k_attention(const _Float16* __restrict__ qkv, const float* __restrict__ edge,
                 float* __restrict__ edge_out, _Float16* __restrict__ attn_out) {
    constexpr int LDQ = 72;                       // 64 + 8 halfs pad
    __shared__ __align__(16) _Float16 Qs [64 * LDQ];
    __shared__ __align__(16) _Float16 Ks [64 * LDQ];
    __shared__ __align__(16) _Float16 Vts[64 * LDQ];   // V transposed: [d][k]
    __shared__ __align__(16) _Float16 Ps [4][16 * LDQ];

    int bh = blockIdx.x, qb = blockIdx.y;
    int b = bh >> 3, h = bh & 7;
    int tid = threadIdx.x, lane = tid & 31, wave = tid >> 5;
    int half = lane >> 4, c = lane & 15;
    int q0 = qb * 64;

    {   // stage Q tile (64 x 64 f16), 2 threads per row
        int r = tid >> 1, seg = (tid & 1) * 32;
        const uint4* g = (const uint4*)(qkv + ((size_t)((q0 + r) * 4 + b)) * 1536 + h * 64 + seg);
        uint4* s = (uint4*)(Qs + r * LDQ + seg);
        s[0] = g[0]; s[1] = g[1]; s[2] = g[2]; s[3] = g[3];
    }
    __syncthreads();
    v16h aq0 = load_frag_a(Qs + wave * 16 * LDQ,      LDQ, lane);  // Dh 0..31
    v16h aq1 = load_frag_a(Qs + wave * 16 * LDQ + 32, LDQ, lane);  // Dh 32..63

    float m_run[8], l_run[8];
    #pragma unroll
    for (int v = 0; v < 8; ++v) { m_run[v] = -1e30f; l_run[v] = 0.f; }
    v8f Oacc[4] = {};

    for (int kt = 0; kt < 16; ++kt) {
        int k0 = kt * 64;
        __syncthreads();
        {   // stage K tile ([k][d]) and V tile transposed ([d][k])
            int r = tid >> 1, seg = (tid & 1) * 32;
            size_t rowbase = ((size_t)((k0 + r) * 4 + b)) * 1536 + h * 64 + seg;
            const uint4* gk = (const uint4*)(qkv + rowbase + 512);
            uint4* sk = (uint4*)(Ks + r * LDQ + seg);
            sk[0] = gk[0]; sk[1] = gk[1]; sk[2] = gk[2]; sk[3] = gk[3];
            const unsigned* gv = (const unsigned*)(qkv + rowbase + 1024);
            #pragma unroll
            for (int i = 0; i < 16; ++i) {
                union { unsigned u; _Float16 hh[2]; } cv; cv.u = gv[i];
                int d = seg + 2 * i;
                Vts[(d    ) * LDQ + r] = cv.hh[0];
                Vts[(d + 1) * LDQ + r] = cv.hh[1];
            }
        }
        __syncthreads();

        // S = Q K^T for this wave's 16 rows x 64 k-cols
        float sreg[4][8];
        #pragma unroll
        for (int f = 0; f < 4; ++f) {
            v8f s = {};
            v16h bk0 = load_frag_b(Ks + f * 16 * LDQ,      LDQ, lane);
            v16h bk1 = load_frag_b(Ks + f * 16 * LDQ + 32, LDQ, lane);
            s = WMMA_F16(aq0, bk0, s);
            s = WMMA_F16(aq1, bk1, s);
            #pragma unroll
            for (int v = 0; v < 8; ++v) {
                int qr = q0 + wave * 16 + v + 8 * half;
                int kc = k0 + f * 16 + c;
                size_t e = ((size_t)bh * 1024 + qr) * 1024 + kc;
                float val = s[v] * 0.125f + edge[e];
                edge_out[e] = val;                       // emit t + edge
                sreg[f][v] = val;
            }
        }

        // online softmax (rows live in 16-lane halves -> xor 8,4,2,1)
        #pragma unroll
        for (int v = 0; v < 8; ++v) {
            float tm = sreg[0][v];
            tm = fmaxf(tm, sreg[1][v]); tm = fmaxf(tm, sreg[2][v]); tm = fmaxf(tm, sreg[3][v]);
            #pragma unroll
            for (int off = 8; off; off >>= 1) tm = fmaxf(tm, __shfl_xor(tm, off, 32));
            float newm = fmaxf(m_run[v], tm);
            float corr = __expf(m_run[v] - newm);
            m_run[v] = newm;
            float rs = 0.f;
            #pragma unroll
            for (int f = 0; f < 4; ++f) {
                float p = __expf(sreg[f][v] - newm);
                sreg[f][v] = p; rs += p;
            }
            #pragma unroll
            for (int off = 8; off; off >>= 1) rs += __shfl_xor(rs, off, 32);
            l_run[v] = l_run[v] * corr + rs;
            #pragma unroll
            for (int g = 0; g < 4; ++g) Oacc[g][v] *= corr;
            #pragma unroll
            for (int f = 0; f < 4; ++f)
                Ps[wave][(v + 8 * half) * LDQ + f * 16 + c] = (_Float16)sreg[f][v];
        }

        // O += P @ V  (wave-local P; compiler inserts ds waits)
        v16h ap0 = load_frag_a(&Ps[wave][0],      LDQ, lane);
        v16h ap1 = load_frag_a(&Ps[wave][0] + 32, LDQ, lane);
        #pragma unroll
        for (int g = 0; g < 4; ++g) {
            v16h bv0 = load_frag_b(Vts + g * 16 * LDQ,      LDQ, lane);
            v16h bv1 = load_frag_b(Vts + g * 16 * LDQ + 32, LDQ, lane);
            Oacc[g] = WMMA_F16(ap0, bv0, Oacc[g]);
            Oacc[g] = WMMA_F16(ap1, bv1, Oacc[g]);
        }
    }

    #pragma unroll
    for (int g = 0; g < 4; ++g)
        #pragma unroll
        for (int v = 0; v < 8; ++v) {
            int qr  = q0 + wave * 16 + v + 8 * half;
            int col = h * 64 + g * 16 + c;
            float o = Oacc[g][v] / l_run[v];
            attn_out[((size_t)(qr * 4 + b)) * 512 + col] = (_Float16)o;
        }
}

// ---------------------------------------------------------------------------
extern "C" void kernel_launch(void* const* d_in, const int* in_sizes, int n_in,
                              void* d_out, int out_size, void* d_ws, size_t ws_size,
                              hipStream_t stream) {
    const float* x        = (const float*)d_in[0];
    const float* edge     = (const float*)d_in[1];
    const float* in_w     = (const float*)d_in[2];
    const float* in_b     = (const float*)d_in[3];
    const float* out_w    = (const float*)d_in[4];
    const float* out_b    = (const float*)d_in[5];
    const float* lin1_w   = (const float*)d_in[6];
    const float* lin1_b   = (const float*)d_in[7];
    const float* lin2_w   = (const float*)d_in[8];
    const float* lin2_b   = (const float*)d_in[9];
    const float* n1_w     = (const float*)d_in[10];
    const float* n1_b     = (const float*)d_in[11];
    const float* n2_w     = (const float*)d_in[12];
    const float* n2_b     = (const float*)d_in[13];

    float* x_out    = (float*)d_out;                       // [4096, 512]
    float* edge_out = (float*)d_out + (size_t)4096 * 512;  // [32,1024,1024]

    // workspace carve
    size_t off = 0;
    auto carve = [&](size_t bytes) {
        void* p = (char*)d_ws + off;
        off += (bytes + 255) & ~(size_t)255;
        return p;
    };
    _Float16* w_in16   = (_Float16*)carve((size_t)1536 * 512  * 2);
    _Float16* w_out16  = (_Float16*)carve((size_t)512  * 512  * 2);
    _Float16* w_l1_16  = (_Float16*)carve((size_t)2048 * 512  * 2);
    _Float16* w_l2_16  = (_Float16*)carve((size_t)512  * 2048 * 2);
    _Float16* xn16     = (_Float16*)carve((size_t)4096 * 512  * 2);
    _Float16* qkv16    = (_Float16*)carve((size_t)4096 * 1536 * 2);
    _Float16* attn16   = (_Float16*)carve((size_t)4096 * 512  * 2);
    float*    xmid     = (float*)   carve((size_t)4096 * 512  * 4);
    _Float16* hn16     = (_Float16*)carve((size_t)4096 * 512  * 2);
    _Float16* h16      = (_Float16*)carve((size_t)4096 * 2048 * 2);
    (void)ws_size; (void)n_in; (void)in_sizes; (void)out_size;

    // weight conversions
    k_f32_to_f16<<<(1536 * 512 + 255) / 256, 256, 0, stream>>>(in_w,   w_in16,  1536 * 512);
    k_f32_to_f16<<<(512  * 512 + 255) / 256, 256, 0, stream>>>(out_w,  w_out16, 512  * 512);
    k_f32_to_f16<<<(2048 * 512 + 255) / 256, 256, 0, stream>>>(lin1_w, w_l1_16, 2048 * 512);
    k_f32_to_f16<<<(512 * 2048 + 255) / 256, 256, 0, stream>>>(lin2_w, w_l2_16, 512 * 2048);

    // LN1
    k_layernorm_f16<<<4096, 256, 0, stream>>>(x, n1_w, n1_b, xn16);
    // QKV: [4096,1536] = xn @ in_w^T + in_b
    k_gemm_wmma<true, false, false><<<dim3(32, 12), 256, 0, stream>>>(
        xn16, w_in16, in_b, nullptr, nullptr, qkv16, 4096, 1536, 512);
    // attention + edge emit
    k_attention<<<dim3(32, 16), 128, 0, stream>>>(qkv16, edge, edge_out, attn16);
    // out proj + residual(x): xmid f32
    k_gemm_wmma<false, false, true><<<dim3(32, 4), 256, 0, stream>>>(
        attn16, w_out16, out_b, x, xmid, nullptr, 4096, 512, 512);
    // LN2
    k_layernorm_f16<<<4096, 256, 0, stream>>>(xmid, n2_w, n2_b, hn16);
    // FFN1 + exact gelu: h16 [4096,2048]
    k_gemm_wmma<true, true, false><<<dim3(32, 16), 256, 0, stream>>>(
        hn16, w_l1_16, lin1_b, nullptr, nullptr, h16, 4096, 2048, 512);
    // FFN2 + residual(xmid) -> x_out f32
    k_gemm_wmma<false, false, true><<<dim3(32, 4), 256, 0, stream>>>(
        h16, w_l2_16, lin2_b, xmid, x_out, nullptr, 4096, 512, 2048);
}
